// MultiHeadAttention_54116587929673
// MI455X (gfx1250) — compile-verified, matrix operand order checked by reading, not run on hardware
//
#include <hip/hip_runtime.h>

typedef __attribute__((ext_vector_type(16))) _Float16 v16h;
typedef __attribute__((ext_vector_type(8)))  float    v8f;

// ---------------------------------------------------------------------------
// Problem constants (fixed by the reference)
// ---------------------------------------------------------------------------
#define BATCH   2
#define SEQ     2048
#define DMODEL  1024
#define NHEADS  16
#define DK      64
#define MROWS   (BATCH * SEQ)      // 4096

// ---------------------------------------------------------------------------
// CDNA5 async global->LDS copy (ASYNCcnt path).  16 bytes per lane, no VGPR
// data movement.  LDS byte address = low 32 bits of the generic pointer
// (flat LDS aperture maps addr[31:0] directly to the wave's LDS offset).
// ---------------------------------------------------------------------------
__device__ __forceinline__ void async_ld16(void* lds_ptr, const void* gptr) {
    const unsigned int       lds = (unsigned int)(uintptr_t)lds_ptr;
    const unsigned long long ga  = (unsigned long long)(uintptr_t)gptr;
    asm volatile("global_load_async_to_lds_b128 %0, %1, off"
                 :: "v"(lds), "v"(ga) : "memory");
}
__device__ __forceinline__ void wait_async0() {
    asm volatile("s_wait_asynccnt 0x0" ::: "memory");
}

// ---------------------------------------------------------------------------
// Fragment gather from a row-major f16 LDS tile (ISA 7.12.2 A-layout).
//   lanes 0-15 : M = lane,    K = {0..7, 16..23}
//   lanes16-31 : M = lane-16, K = {8..15, 24..31}
// Identical gather serves B-fragments when the tile holds B^T rows.
// Compiler coalesces this into two ds_load_b128 per fragment.
// ---------------------------------------------------------------------------
__device__ __forceinline__ v16h load_frag(const _Float16* __restrict__ base,
                                          int ld, int lane) {
    const int r     = lane & 15;
    const int khalf = (lane >> 4) * 8;
    const _Float16* row = base + r * ld;
    v16h f;
#pragma unroll
    for (int p = 0; p < 8; ++p) {
        const int kk = ((p & 4) ? 16 : 0) + khalf + (p & 3) * 2;
        f[2 * p]     = row[kk];
        f[2 * p + 1] = row[kk + 1];
    }
    return f;
}

__device__ __forceinline__ v8f wmma_f16(v16h a, v16h b, v8f c) {
    return __builtin_amdgcn_wmma_f32_16x16x32_f16(
        false, a, false, b, (short)0, c, false, false);
}

// ---------------------------------------------------------------------------
// f32 -> f16 conversion pass (bandwidth-bound, done once so the GEMM hot
// loops carry zero conversion VALU).
// ---------------------------------------------------------------------------
__global__ __launch_bounds__(256)
void cvt_kernel(const float* __restrict__ in, _Float16* __restrict__ out) {
    const size_t i = ((size_t)blockIdx.x * blockDim.x + threadIdx.x) * 4;
    const float4 f = *(const float4*)(in + i);
    out[i + 0] = (_Float16)f.x;
    out[i + 1] = (_Float16)f.y;
    out[i + 2] = (_Float16)f.z;
    out[i + 3] = (_Float16)f.w;
}

// ---------------------------------------------------------------------------
// GEMM: C[M,N] = A[M,K] @ W[N,K]^T + bias[N]   (A, W already f16)
//   SCAT : scatter f16 output into [B, H, S, DK] head-major (with scale)
//          else write f32 row-major [M, N]
// Block: 256 threads = 8 wave32s (2M x 4N); block tile 64x128, wave tile
// 32x32 (2x2 fragments), K step 32.  Double-buffered async-to-LDS pipeline:
//   iter i:  s_wait_asynccnt 0 (tile i landed) ; barrier ;
//            issue async tile i+1 ; WMMA on tile i.
// ---------------------------------------------------------------------------
#define BM 64
#define BN 128
#define BK 32
#define LDT (BK + 8)   // 40 halves = 80 B row stride (16 B aligned)

template <bool SCAT>
__global__ __launch_bounds__(256)
void gemm_kernel(const _Float16* __restrict__ A,
                 const _Float16* __restrict__ W,
                 const float* __restrict__ bias,
                 void* __restrict__ outv,
                 float scale) {
    __shared__ _Float16 As[2][BM * LDT];
    __shared__ _Float16 Bs[2][BN * LDT];

    const int tid  = threadIdx.x;
    const int lane = tid & 31;
    const int wid  = tid >> 5;
    const int bm   = blockIdx.y * BM;
    const int bn   = blockIdx.x * BN;
    const int wm   = (wid >> 2) * 32;
    const int wn   = (wid & 3) * 32;

    v8f acc[2][2];
#pragma unroll
    for (int i = 0; i < 2; ++i)
#pragma unroll
        for (int j = 0; j < 2; ++j) acc[i][j] = (v8f){};

    // async staging assignment: A 64x32 halves (1 op/thread),
    //                           B 128x32 halves (2 ops/thread)
    const int arow = tid >> 2, acol = (tid & 3) * 8;
    const int brow = tid >> 1, bcol = (tid & 1) * 16;

    auto stage = [&](int k0, int buf) {
        async_ld16(&As[buf][arow * LDT + acol],
                   A + (size_t)(bm + arow) * DMODEL + k0 + acol);
        const _Float16* wg = W + (size_t)(bn + brow) * DMODEL + k0 + bcol;
        async_ld16(&Bs[buf][brow * LDT + bcol],     wg);
        async_ld16(&Bs[buf][brow * LDT + bcol + 8], wg + 8);
    };

    stage(0, 0);
    int buf = 0;
    for (int k0 = 0; k0 < DMODEL; k0 += BK) {
        wait_async0();          // my share of tile k0 is in LDS
        __syncthreads();        // everyone's share is in LDS
        if (k0 + BK < DMODEL) stage(k0 + BK, buf ^ 1);   // prefetch next tile

        v16h af[2], bf[2];
        af[0] = load_frag(&As[buf][(wm + 0)  * LDT], LDT, lane);
        af[1] = load_frag(&As[buf][(wm + 16) * LDT], LDT, lane);
        bf[0] = load_frag(&Bs[buf][(wn + 0)  * LDT], LDT, lane);
        bf[1] = load_frag(&Bs[buf][(wn + 16) * LDT], LDT, lane);

#pragma unroll
        for (int mi = 0; mi < 2; ++mi)
#pragma unroll
            for (int ni = 0; ni < 2; ++ni)
                acc[mi][ni] = wmma_f16(af[mi], bf[ni], acc[mi][ni]);
        buf ^= 1;
    }

    // epilogue.  C/D layout: VGPR r -> M = r + 8*(lane>=16), N = lane&15
    const int rbase = (lane >> 4) * 8;
    const int cn    = lane & 15;
#pragma unroll
    for (int mi = 0; mi < 2; ++mi) {
#pragma unroll
        for (int ni = 0; ni < 2; ++ni) {
#pragma unroll
            for (int r = 0; r < 8; ++r) {
                const int gm = bm + wm + mi * 16 + r + rbase;
                const int gn = bn + wn + ni * 16 + cn;
                float val = acc[mi][ni][r] + bias[gn];
                if (SCAT) {
                    const int b = gm >> 11, s = gm & (SEQ - 1);
                    const int h = gn >> 6,  d = gn & (DK - 1);
                    ((_Float16*)outv)[(((size_t)(b * NHEADS + h) * SEQ) + s) * DK + d] =
                        (_Float16)(val * scale);
                } else {
                    ((float*)outv)[(size_t)gm * DMODEL + gn] = val;
                }
            }
        }
    }
}

// ---------------------------------------------------------------------------
// Flash attention over one (batch, head, 64-query tile).
// Block: 128 threads = 4 wave32s; each wave owns 16 query rows.
// Q tile async-staged once; K tiles async-staged per step; V staged
// transposed (16-bit scatter, so register path).
// Row max: 16-lane xor-shuffle reduction.
// Row sum: computed as P @ ones with two WMMAs -- the D-layout returns each
// row's sum replicated across its 16-lane half, removing the bpermute/add
// reduction tree entirely and moving the work to the matrix pipe.
// ---------------------------------------------------------------------------
#define LT 72   // 144 B row stride (16 B aligned)

__global__ __launch_bounds__(128)
void attn_kernel(const _Float16* __restrict__ Qh,
                 const _Float16* __restrict__ Kh,
                 const _Float16* __restrict__ Vh,
                 const int* __restrict__ mask,
                 _Float16* __restrict__ Ctx) {
    __shared__ _Float16 Qs[64 * LT];
    __shared__ _Float16 Ks[64 * LT];
    __shared__ _Float16 Vs[64 * LT];       // transposed: Vs[d*LT + krow]
    __shared__ _Float16 Ps[4][16 * LT];    // per-wave P staging

    const int tid  = threadIdx.x;
    const int lane = tid & 31;
    const int wid  = tid >> 5;
    const int qbase = blockIdx.x * 64;
    const int h     = blockIdx.y;
    const int b     = blockIdx.z;

    const size_t headOff = (size_t)(b * NHEADS + h) * SEQ * DK;
    const _Float16* Qp = Qh + headOff;
    const _Float16* Kp = Kh + headOff;
    const _Float16* Vp = Vh + headOff;

    const int srow = tid >> 1;             // 0..63
    const int scol = (tid & 1) * 32;       // 0 / 32

    // ---- stage Q tile (async, 4 x 16B per thread) ----
    {
        const _Float16* src = Qp + (size_t)(qbase + srow) * DK + scol;
#pragma unroll
        for (int i = 0; i < 4; ++i)
            async_ld16(&Qs[srow * LT + scol + i * 8], src + i * 8);
    }
    wait_async0();
    __syncthreads();

    const int mrow = wid * 16;
    const v16h aq0 = load_frag(&Qs[mrow * LT + 0],  LT, lane);
    const v16h aq1 = load_frag(&Qs[mrow * LT + 32], LT, lane);

    // constant all-ones B fragment for the row-sum WMMA
    v16h ones;
#pragma unroll
    for (int i = 0; i < 16; ++i) ones[i] = (_Float16)1.0f;

    float rowmax[8], rowsum[8];
#pragma unroll
    for (int r = 0; r < 8; ++r) { rowmax[r] = -3.0e38f; rowsum[r] = 0.0f; }
    v8f co[4];
#pragma unroll
    for (int oi = 0; oi < 4; ++oi) co[oi] = (v8f){};

    const int rbase = (lane >> 4) * 8;
    const int cn    = lane & 15;

    for (int kv = 0; kv < SEQ; kv += 64) {
        __syncthreads();
        // ---- K tile: async to LDS ----
        {
            const _Float16* ksrc = Kp + (size_t)(kv + srow) * DK + scol;
#pragma unroll
            for (int i = 0; i < 4; ++i)
                async_ld16(&Ks[srow * LT + scol + i * 8], ksrc + i * 8);
        }
        // ---- V tile: register transpose (16-bit scatter) ----
        {
            const _Float16* vsrc = Vp + (size_t)(kv + srow) * DK + scol;
#pragma unroll
            for (int i = 0; i < 32; ++i) Vs[(scol + i) * LT + srow] = vsrc[i];
        }
        wait_async0();
        __syncthreads();

        // ---- scores: S = Q @ K^T  (Q already carries 1/sqrt(dk)) ----
        v8f sc[4];
#pragma unroll
        for (int ni = 0; ni < 4; ++ni) {
            v16h bk0 = load_frag(&Ks[(ni * 16) * LT + 0],  LT, lane);
            v16h bk1 = load_frag(&Ks[(ni * 16) * LT + 32], LT, lane);
            v8f z = (v8f){};
            z = wmma_f16(aq0, bk0, z);
            z = wmma_f16(aq1, bk1, z);
            sc[ni] = z;
        }

        // ---- apply mask (+prefetch next tile's mask rows) ----
        const int* mbase = mask + ((size_t)b * SEQ + (qbase + mrow)) * SEQ + kv;
        if (kv + 64 < SEQ) __builtin_prefetch(mbase + 64, 0, 0);
#pragma unroll
        for (int ni = 0; ni < 4; ++ni) {
#pragma unroll
            for (int r = 0; r < 8; ++r) {
                if (mbase[(size_t)(r + rbase) * SEQ + ni * 16 + cn] == 0)
                    sc[ni][r] = -1.0e9f;
            }
        }

        // ---- online softmax: max via 16-lane shuffles, exp, rescale ----
#pragma unroll
        for (int r = 0; r < 8; ++r) {
            float mx = sc[0][r];
#pragma unroll
            for (int ni = 1; ni < 4; ++ni) mx = fmaxf(mx, sc[ni][r]);
#pragma unroll
            for (int off = 1; off < 16; off <<= 1)
                mx = fmaxf(mx, __shfl_xor(mx, off, 32));
            const float mnew = fmaxf(rowmax[r], mx);
            const float corr = __expf(rowmax[r] - mnew);
            rowmax[r] = mnew;
            rowsum[r] *= corr;
#pragma unroll
            for (int ni = 0; ni < 4; ++ni)
                sc[ni][r] = __expf(sc[ni][r] - mnew);
#pragma unroll
            for (int oi = 0; oi < 4; ++oi) co[oi][r] *= corr;
        }

        // ---- C-layout -> A-layout via per-wave LDS ----
        _Float16* pw = Ps[wid];
#pragma unroll
        for (int ni = 0; ni < 4; ++ni)
#pragma unroll
            for (int r = 0; r < 8; ++r)
                pw[(r + rbase) * LT + ni * 16 + cn] = (_Float16)sc[ni][r];
        // per-wave staging area: same-wave LDS ordering (DScnt) suffices
        const v16h ap0 = load_frag(&pw[0],  LT, lane);
        const v16h ap1 = load_frag(&pw[32], LT, lane);

        // ---- row sums via WMMA: P @ ones (replicated per 16-lane half) ----
        {
            v8f z = (v8f){};
            z = wmma_f16(ap0, ones, z);
            z = wmma_f16(ap1, ones, z);
#pragma unroll
            for (int r = 0; r < 8; ++r) rowsum[r] += z[r];
        }

        // ---- ctx += P @ V ----
#pragma unroll
        for (int oi = 0; oi < 4; ++oi) {
            v16h bv0 = load_frag(&Vs[(oi * 16) * LT + 0],  LT, lane);
            v16h bv1 = load_frag(&Vs[(oi * 16) * LT + 32], LT, lane);
            co[oi] = wmma_f16(ap0, bv0, co[oi]);
            co[oi] = wmma_f16(ap1, bv1, co[oi]);
        }
    }

    // ---- normalize and store context as f16 [B, S, H*DK] ----
#pragma unroll
    for (int oi = 0; oi < 4; ++oi) {
#pragma unroll
        for (int r = 0; r < 8; ++r) {
            const int s = qbase + mrow + r + rbase;
            const int d = oi * 16 + cn;
            const float val = co[oi][r] / rowsum[r];
            Ctx[((size_t)(b * SEQ + s) * DMODEL) + h * DK + d] = (_Float16)val;
        }
    }
}

// ---------------------------------------------------------------------------
// Launcher
// inputs: query, key, value, mask, Wq, bq, Wk, bk, Wv, bv, Wo, bo
// ws (f16 halves): xq|xk|xv (3x4M) | w16 (4x1M) | Qh|Kh|Vh (3x4M) | ctx (4M)
//                  = 32M halves = 64 MB
// ---------------------------------------------------------------------------
extern "C" void kernel_launch(void* const* d_in, const int* in_sizes, int n_in,
                              void* d_out, int out_size, void* d_ws, size_t ws_size,
                              hipStream_t stream) {
    (void)in_sizes; (void)n_in; (void)out_size; (void)ws_size;

    const float* q    = (const float*)d_in[0];
    const float* k    = (const float*)d_in[1];
    const float* v    = (const float*)d_in[2];
    const int*   mask = (const int*)d_in[3];
    const float* Wq   = (const float*)d_in[4];
    const float* bq   = (const float*)d_in[5];
    const float* Wk   = (const float*)d_in[6];
    const float* bk   = (const float*)d_in[7];
    const float* Wv   = (const float*)d_in[8];
    const float* bv   = (const float*)d_in[9];
    const float* Wo   = (const float*)d_in[10];
    const float* bo   = (const float*)d_in[11];
    float* out = (float*)d_out;

    const size_t nX = (size_t)MROWS * DMODEL;      // 4,194,304
    const size_t nW = (size_t)DMODEL * DMODEL;     // 1,048,576

    _Float16* xq   = (_Float16*)d_ws;
    _Float16* xk   = xq + nX;
    _Float16* xv   = xk + nX;
    _Float16* w16q = xv + nX;
    _Float16* w16k = w16q + nW;
    _Float16* w16v = w16k + nW;
    _Float16* w16o = w16v + nW;
    _Float16* qh   = w16o + nW;
    _Float16* kh   = qh + nX;
    _Float16* vh   = kh + nX;
    _Float16* ctx  = vh + nX;

    // one-time f32 -> f16 conversion (bandwidth-trivial)
    cvt_kernel<<<dim3(nX / (256 * 4)), 256, 0, stream>>>(q, xq);
    cvt_kernel<<<dim3(nX / (256 * 4)), 256, 0, stream>>>(k, xk);
    cvt_kernel<<<dim3(nX / (256 * 4)), 256, 0, stream>>>(v, xv);
    cvt_kernel<<<dim3(nW / (256 * 4)), 256, 0, stream>>>(Wq, w16q);
    cvt_kernel<<<dim3(nW / (256 * 4)), 256, 0, stream>>>(Wk, w16k);
    cvt_kernel<<<dim3(nW / (256 * 4)), 256, 0, stream>>>(Wv, w16v);
    cvt_kernel<<<dim3(nW / (256 * 4)), 256, 0, stream>>>(Wo, w16o);

    const dim3 ggrid(DMODEL / BN, MROWS / BM);   // (8, 64)
    const dim3 gblk(256);
    const float qscale = 0.125f;                  // 1/sqrt(DK)

    gemm_kernel<true><<<ggrid, gblk, 0, stream>>>(xq, w16q, bq, qh, qscale);
    gemm_kernel<true><<<ggrid, gblk, 0, stream>>>(xk, w16k, bk, kh, 1.0f);
    gemm_kernel<true><<<ggrid, gblk, 0, stream>>>(xv, w16v, bv, vh, 1.0f);

    attn_kernel<<<dim3(SEQ / 64, NHEADS, BATCH), 128, 0, stream>>>(qh, kh, vh, mask, ctx);

    gemm_kernel<false><<<ggrid, gblk, 0, stream>>>(ctx, w16o, bo, out, 1.0f);
}
